// RQAutoEncoder_38225208934990
// MI455X (gfx1250) — matrix operand order repaired and checked by reading, not compiled, not run on hardware
//
#include <hip/hip_runtime.h>

#define NROWS 16384
#define EMB   4096
#define DD1   512
#define DD2   256
#define DIM   32
#define NQ    3
#define CBN   7000
#define CBP   7008
#define VQ_TILES  (CBP / 16)     // 438
#define VQ_BLOCKS (NROWS / 16)   // 1024

typedef __attribute__((ext_vector_type(16))) _Float16 v16h;
typedef __attribute__((ext_vector_type(8)))  float    v8f;

struct alignas(16) Bytes16 { unsigned int u[4]; };

union F16Frag {
  v16h     v;
  _Float16 h[16];
  Bytes16  b[2];
};

// ---- Fragment loaders matching CDNA5 WMMA VGPR layouts (CDNA5 ISA) ----

// A: 16x32 f16. lane&15 = row m; lanes<16 hold K[0..7],K[16..23]; lanes>=16 hold K[8..15],K[24..31].
__device__ inline v16h load_a_f16(const _Float16* A, int lda, int row0, int kbase, int lane) {
  int m    = row0 + (lane & 15);
  int koff = (lane & 16) >> 1;                       // 0 or 8
  const _Float16* p = A + (size_t)m * lda + kbase + koff;
  F16Frag r;
  r.b[0] = *(const Bytes16*)p;                       // K = koff .. koff+7
  r.b[1] = *(const Bytes16*)(p + 16);                // K = 16+koff .. 16+koff+7
  return r.v;
}

__device__ inline v16h load_a_f32(const float* A, int lda, int row0, int kbase, int lane) {
  int m    = row0 + (lane & 15);
  int koff = (lane & 16) >> 1;
  const float* p0 = A + (size_t)m * lda + kbase + koff;
  const float* p1 = p0 + 16;
  F16Frag r;
#pragma unroll
  for (int e = 0; e < 8; ++e) { r.h[e] = (_Float16)p0[e]; r.h[8 + e] = (_Float16)p1[e]; }
  return r.v;
}

// B: 32x16 f16 from transposed weight Wt[N][K]. lanes<16: K[0..7],K[16..23]; lanes>=16: K[8..15],K[24..31]; n = lane&15.
__device__ inline v16h load_bt(const _Float16* Bt, int K, int n, int kbase, int lane) {
  int klo = (lane & 16) >> 1;                        // 0 or 8
  const _Float16* p = Bt + (size_t)n * K + kbase + klo;
  F16Frag r;
  r.b[0] = *(const Bytes16*)p;
  r.b[1] = *(const Bytes16*)(p + 16);
  return r.v;
}

#define WMMA_F16(a, b, c) \
  __builtin_amdgcn_wmma_f32_16x16x32_f16(false, (a), false, (b), (short)0, (c), false, false)

// ---- Generic fused GEMM: C = act(A @ Wt^T + bias) ----
// 256 threads = 8 waves as 4 row-groups x 2 col-groups; block tile = 128 rows x 128 cols.
// Each wave: 32 rows x 64 cols -> 8 accumulators, 8 v_wmma per K-step.
template <int AF32, int RELU, int OUTF16>
__global__ __launch_bounds__(256) void gemm_kernel(
    const void* __restrict__ Av, const _Float16* __restrict__ Bt,
    const float* __restrict__ bias, _Float16* __restrict__ Ch,
    float* __restrict__ Cf, int N, int K) {
  int lane   = threadIdx.x & 31;
  int wave   = threadIdx.x >> 5;
  int rowgrp = wave >> 1;
  int colgrp = wave & 1;
  int row0 = blockIdx.y * 128 + rowgrp * 32;
  int col0 = blockIdx.x * 128 + colgrp * 64;

  v8f acc[2][4];
#pragma unroll
  for (int i = 0; i < 2; ++i)
#pragma unroll
    for (int j = 0; j < 4; ++j) acc[i][j] = (v8f){};

  for (int kbase = 0; kbase < K; kbase += 32) {
    v16h a0, a1;
    if (AF32) {
      const float* A = (const float*)Av;
      a0 = load_a_f32(A, K, row0, kbase, lane);
      a1 = load_a_f32(A, K, row0 + 16, kbase, lane);
      // speculative prefetch of next K tile (invalid translations are dropped)
      __builtin_prefetch(A + (size_t)(row0 + (lane & 15)) * K + kbase + 32, 0, 1);
    } else {
      const _Float16* A = (const _Float16*)Av;
      a0 = load_a_f16(A, K, row0, kbase, lane);
      a1 = load_a_f16(A, K, row0 + 16, kbase, lane);
      __builtin_prefetch(A + (size_t)(row0 + (lane & 15)) * K + kbase + 32, 0, 1);
    }
#pragma unroll
    for (int ct = 0; ct < 4; ++ct) {
      int n = col0 + ct * 16 + (lane & 15);
      if (n >= N) n = N - 1;                         // clamp (guarded at store)
      v16h b = load_bt(Bt, K, n, kbase, lane);
      acc[0][ct] = WMMA_F16(a0, b, acc[0][ct]);
      acc[1][ct] = WMMA_F16(a1, b, acc[1][ct]);
    }
  }

  // D layout: VGPR r -> M = r (lanes 0-15) or r+8 (lanes 16-31); N = lane&15
  int mhi = (lane & 16) >> 1;
#pragma unroll
  for (int rt = 0; rt < 2; ++rt) {
#pragma unroll
    for (int ct = 0; ct < 4; ++ct) {
      int n = col0 + ct * 16 + (lane & 15);
      if (n < N) {
        float bv = bias[n];
#pragma unroll
        for (int r = 0; r < 8; ++r) {
          int m = row0 + rt * 16 + mhi + r;
          float v = acc[rt][ct][r] + bv;
          if (RELU) v = fmaxf(v, 0.0f);
          if (OUTF16) Ch[(size_t)m * N + n] = (_Float16)v;
          else        Cf[(size_t)m * N + n] = v;
        }
      }
    }
  }
}

// ---- One residual-VQ stage: WMMA scores + argmin + gather + residual update + loss partial ----
__global__ __launch_bounds__(32) void vq_step_kernel(
    const _Float16* __restrict__ cbh_all, const float* __restrict__ cnorm_all,
    const float* __restrict__ cbf_all, float* __restrict__ residual,
    float* __restrict__ quant_out, float* __restrict__ out_idx,
    float* __restrict__ loss_partial, int q) {
  int lane = threadIdx.x;
  int rowbase = blockIdx.x * 16;
  __shared__ int sidx[16];

  v16h a = load_a_f32(residual, DIM, rowbase, 0, lane);   // 16 rows x K=32, single frag

  const _Float16* cbh = cbh_all + (size_t)q * CBP * DIM;
  const float*    cn  = cnorm_all + (size_t)q * CBP;

  float minv[8];
  int   mini[8];
#pragma unroll
  for (int r = 0; r < 8; ++r) { minv[r] = __builtin_inff(); mini[r] = 0; }

  int nlo = lane & 15;
  int klo = lane & 16;
  for (int t = 0; t < VQ_TILES; ++t) {
    int col = t * 16 + nlo;
    // unconditional speculative prefetch ~4 tiles ahead (OOB prefetch is dropped)
    __builtin_prefetch(cbh + (size_t)(col + 64) * DIM + klo, 0, 1);
    v16h b = load_bt(cbh, DIM, col, 0, lane);
    v8f  c = (v8f){};
    c = WMMA_F16(a, b, c);
    float cnv = cn[col];
#pragma unroll
    for (int r = 0; r < 8; ++r) {
      float sc = __builtin_fmaf(-2.0f, c[r], cnv);        // ||c||^2 - 2 r.c
      if (sc < minv[r]) { minv[r] = sc; mini[r] = col; }  // keep earliest tile on ties
    }
  }

  // cross-lane argmin within each 16-lane half (lanes 0-15: rows r; lanes 16-31: rows r+8)
#pragma unroll
  for (int r = 0; r < 8; ++r) {
    float v = minv[r];
    int   i = mini[r];
#pragma unroll
    for (int mask = 8; mask >= 1; mask >>= 1) {
      float ov = __shfl_xor(v, mask, 32);
      int   oi = __shfl_xor(i, mask, 32);
      if (ov < v || (ov == v && oi < i)) { v = ov; i = oi; }
    }
    if ((lane & 15) == 0) sidx[r + ((lane >> 4) << 3)] = i;
  }
  __syncthreads();

  // gather winning codeword (f32), update residual & quant_out, accumulate loss
  const float* cbf = cbf_all + (size_t)q * CBN * DIM;
  float lsum = 0.0f;
#pragma unroll 4
  for (int rr = 0; rr < 16; ++rr) {
    int row = rowbase + rr;
    int idx = sidx[rr];
    float  qv = cbf[(size_t)idx * DIM + lane];            // lane = dim
    size_t off = (size_t)row * DIM + lane;
    float  rv = residual[off];
    float  nr = rv - qv;
    residual[off]  = nr;
    quant_out[off] += qv;
    lsum += nr * nr;                                      // (quant - residual)^2 == nr^2
    if (lane == 0) out_idx[(size_t)row * NQ + q] = (float)idx;
  }
#pragma unroll
  for (int mask = 16; mask >= 1; mask >>= 1) lsum += __shfl_xor(lsum, mask, 32);
  if (lane == 0) loss_partial[q * VQ_BLOCKS + blockIdx.x] = lsum;
}

// ---- Small helper kernels ----
__global__ void transpose_w_kernel(const float* __restrict__ W, _Float16* __restrict__ Wt,
                                   int K, int N) {
  int k = blockIdx.x * 32 + threadIdx.x;
  int n = blockIdx.y * 8 + threadIdx.y;
  if (k < K && n < N) Wt[(size_t)n * K + k] = (_Float16)W[(size_t)k * N + n];
}

__global__ __launch_bounds__(32) void prep_cb_kernel(const float* __restrict__ cb,
                                                     _Float16* __restrict__ cbh,
                                                     float* __restrict__ cnorm) {
  int row = blockIdx.x;      // 0..CBP-1 (padded)
  int q   = blockIdx.y;
  int lane = threadIdx.x;    // = dim
  float v = 0.0f;
  if (row < CBN) v = cb[((size_t)q * CBN + row) * DIM + lane];
  cbh[((size_t)q * CBP + row) * DIM + lane] = (_Float16)v;
  float s = v * v;
#pragma unroll
  for (int m = 16; m >= 1; m >>= 1) s += __shfl_xor(s, m, 32);
  if (lane == 0) cnorm[(size_t)q * CBP + row] = (row < CBN) ? s : __builtin_inff();
}

__global__ void zero_kernel(float* __restrict__ p, int n) {
  int i = blockIdx.x * blockDim.x + threadIdx.x;
  if (i < n) p[i] = 0.0f;
}

__global__ void cvt_f32_to_f16_kernel(const float* __restrict__ a, _Float16* __restrict__ b, int n) {
  int i = blockIdx.x * blockDim.x + threadIdx.x;
  if (i < n) b[i] = (_Float16)a[i];
}

__global__ void loss_fin_kernel(const float* __restrict__ part, float* __restrict__ out) {
  int q = threadIdx.x;
  if (q < NQ) {
    float s = 0.0f;
    for (int i = 0; i < VQ_BLOCKS; ++i) s += part[q * VQ_BLOCKS + i];  // fixed order: deterministic
    out[q] = s * (1.0f / ((float)NROWS * (float)DIM));
  }
}

extern "C" void kernel_launch(void* const* d_in, const int* in_sizes, int n_in,
                              void* d_out, int out_size, void* d_ws, size_t ws_size,
                              hipStream_t stream) {
  (void)in_sizes; (void)n_in; (void)out_size; (void)ws_size;
  const float* x    = (const float*)d_in[0];
  const float* ew0  = (const float*)d_in[1];
  const float* eb0  = (const float*)d_in[2];
  const float* ew1  = (const float*)d_in[3];
  const float* eb1  = (const float*)d_in[4];
  const float* ew2  = (const float*)d_in[5];
  const float* eb2  = (const float*)d_in[6];
  const float* cbks = (const float*)d_in[7];
  const float* dw0  = (const float*)d_in[8];
  const float* db0  = (const float*)d_in[9];
  const float* dw1  = (const float*)d_in[10];
  const float* db1  = (const float*)d_in[11];
  const float* dw2  = (const float*)d_in[12];
  const float* db2  = (const float*)d_in[13];

  char* ws = (char*)d_ws;
  size_t off = 0;
  auto alloc = [&](size_t bytes) -> void* {
    void* p = ws + off;
    off = (off + bytes + 255) & ~(size_t)255;
    return p;
  };

  _Float16* act0  = (_Float16*)alloc((size_t)NROWS * DD1 * 2);
  _Float16* act1  = (_Float16*)alloc((size_t)NROWS * DD1 * 2);
  _Float16* w0t   = (_Float16*)alloc((size_t)EMB * DD1 * 2);
  _Float16* w1t   = (_Float16*)alloc((size_t)DD1 * DD2 * 2);
  _Float16* w2t   = (_Float16*)alloc((size_t)DD2 * DIM * 2);
  _Float16* d0t   = (_Float16*)alloc((size_t)DIM * DD2 * 2);
  _Float16* d1t   = (_Float16*)alloc((size_t)DD2 * DD1 * 2);
  _Float16* d2t   = (_Float16*)alloc((size_t)DD1 * EMB * 2);
  _Float16* cbh   = (_Float16*)alloc((size_t)NQ * CBP * DIM * 2);
  float* cnorm    = (float*)alloc((size_t)NQ * CBP * 4);
  float* residual = (float*)alloc((size_t)NROWS * DIM * 4);
  float* quant    = (float*)alloc((size_t)NROWS * DIM * 4);
  _Float16* qh    = (_Float16*)alloc((size_t)NROWS * DIM * 2);
  float* lpart    = (float*)alloc((size_t)NQ * VQ_BLOCKS * 4);

  float* out      = (float*)d_out;
  float* out_idx  = out + (size_t)NROWS * EMB;
  float* out_loss = out_idx + (size_t)NROWS * NQ;

  // weight prep
  dim3 tb(32, 8);
  transpose_w_kernel<<<dim3((EMB + 31) / 32, (DD1 + 7) / 8), tb, 0, stream>>>(ew0, w0t, EMB, DD1);
  transpose_w_kernel<<<dim3((DD1 + 31) / 32, (DD2 + 7) / 8), tb, 0, stream>>>(ew1, w1t, DD1, DD2);
  transpose_w_kernel<<<dim3((DD2 + 31) / 32, (DIM + 7) / 8), tb, 0, stream>>>(ew2, w2t, DD2, DIM);
  transpose_w_kernel<<<dim3((DIM + 31) / 32, (DD2 + 7) / 8), tb, 0, stream>>>(dw0, d0t, DIM, DD2);
  transpose_w_kernel<<<dim3((DD2 + 31) / 32, (DD1 + 7) / 8), tb, 0, stream>>>(dw1, d1t, DD2, DD1);
  transpose_w_kernel<<<dim3((DD1 + 31) / 32, (EMB + 7) / 8), tb, 0, stream>>>(dw2, d2t, DD1, EMB);
  prep_cb_kernel<<<dim3(CBP, NQ), dim3(32), 0, stream>>>(cbks, cbh, cnorm);
  zero_kernel<<<(NROWS * DIM + 255) / 256, 256, 0, stream>>>(quant, NROWS * DIM);

  dim3 blk(256);
  // encoder
  gemm_kernel<1, 1, 1><<<dim3((DD1 + 127) / 128, NROWS / 128), blk, 0, stream>>>(x, w0t, eb0, act0, nullptr, DD1, EMB);
  gemm_kernel<0, 1, 1><<<dim3((DD2 + 127) / 128, NROWS / 128), blk, 0, stream>>>(act0, w1t, eb1, act1, nullptr, DD2, DD1);
  gemm_kernel<0, 0, 0><<<dim3(1, NROWS / 128), blk, 0, stream>>>(act1, w2t, eb2, nullptr, residual, DIM, DD2);

  // residual VQ (3 dependent stages)
  for (int q = 0; q < NQ; ++q)
    vq_step_kernel<<<VQ_BLOCKS, 32, 0, stream>>>(cbh, cnorm, cbks, residual, quant, out_idx, lpart, q);

  cvt_f32_to_f16_kernel<<<(NROWS * DIM + 255) / 256, 256, 0, stream>>>(quant, qh, NROWS * DIM);

  // decoder
  gemm_kernel<0, 1, 1><<<dim3((DD2 + 127) / 128, NROWS / 128), blk, 0, stream>>>(qh, d0t, db0, act1, nullptr, DD2, DIM);
  gemm_kernel<0, 1, 1><<<dim3((DD1 + 127) / 128, NROWS / 128), blk, 0, stream>>>(act1, d1t, db1, act0, nullptr, DD1, DD2);
  gemm_kernel<0, 0, 0><<<dim3(EMB / 128, NROWS / 128), blk, 0, stream>>>(act0, d2t, db2, nullptr, out, EMB, DD1);

  loss_fin_kernel<<<1, 32, 0, stream>>>(lpart, out_loss);
}
